// CNNCharEncode_84688165143160
// MI455X (gfx1250) — matrix operand order repaired and checked by reading, not compiled, not run on hardware
//
#include <hip/hip_runtime.h>
#include <hip/hip_bf16.h>

typedef __attribute__((ext_vector_type(16))) _Float16 v16h;
typedef __attribute__((ext_vector_type(8)))  _Float16 v8h;
typedef __attribute__((ext_vector_type(8)))  float    v8f;

#define WSTRIDE 1056   // halves per word slot in LDS (1000 data + >=56 zero pad)
#define OUT_DIM 768
#define NWORDS  4096

// ---------------------------------------------------------------------------
// A-fragment loader (16x32 f16). Rows = M, contiguous K order (same as B):
//   lanes 0-15:  M=lane,    vgpr0-3 = K 0..7,  vgpr4-7 = K 8..15
//   lanes 16-31: M=lane-16, vgpr0-3 = K 16..23, vgpr4-7 = K 24..31
// Caller passes the per-lane row pointer already offset by (K base = hi*16).
// Only 4-byte alignment is guaranteed, so assemble from dword loads.
// ---------------------------------------------------------------------------
__device__ __forceinline__ v16h load_a16(const _Float16* p) {
    union { v16h h; unsigned int u[8]; } U;
    const unsigned int* lo = (const unsigned int*)p;         // K +0..7
    const unsigned int* hi = (const unsigned int*)(p + 8);   // K +8..15
#pragma unroll
    for (int i = 0; i < 4; ++i) { U.u[i] = lo[i]; U.u[4 + i] = hi[i]; }
    return U.h;
}

// ---------------------------------------------------------------------------
// Kernel 0: weight prep.
//  - conv weights (h,50,w) fp32 -> f16 [o][K_pad], K index k = t*50+i,
//    zero padded for k >= 50*w (so A-side over-reads multiply by 0).
//  - highway weights (o,i) fp32 -> f16 same layout (K=i contiguous, 768%32==0)
// ---------------------------------------------------------------------------
__device__ __forceinline__ void conv_pack(const float* __restrict__ src,
                                          _Float16* __restrict__ dst,
                                          int i, int wd, int Kpad) {
    int o = i / Kpad, k = i - o * Kpad;
    float v = 0.0f;
    if (k < 50 * wd) {
        int t = k / 50, ci = k - t * 50;
        v = src[(o * 50 + ci) * wd + t];
    }
    dst[i] = (_Float16)v;
}

__global__ void __launch_bounds__(256) prep_kernel(
    const float* __restrict__ w0, const float* __restrict__ w1,
    const float* __restrict__ w2, const float* __restrict__ whw,
    const float* __restrict__ wtw,
    _Float16* __restrict__ d_wt0, _Float16* __restrict__ d_wt1,
    _Float16* __restrict__ d_wt2, _Float16* __restrict__ d_whw,
    _Float16* __restrict__ d_wtw) {
    const int N0 = 128 * 64, N1 = 128 * 128, N2 = 512 * 160, NH = OUT_DIM * OUT_DIM;
    const int total = N0 + N1 + N2 + 2 * NH;
    for (int idx = blockIdx.x * blockDim.x + threadIdx.x; idx < total;
         idx += gridDim.x * blockDim.x) {
        int i = idx;
        if (i < N0) { conv_pack(w0, d_wt0, i, 1, 64); continue; }
        i -= N0;
        if (i < N1) { conv_pack(w1, d_wt1, i, 2, 128); continue; }
        i -= N1;
        if (i < N2) { conv_pack(w2, d_wt2, i, 3, 160); continue; }
        i -= N2;
        if (i < NH) { d_whw[i] = (_Float16)whw[i]; continue; }
        i -= NH;
        d_wtw[i] = (_Float16)wtw[i];
    }
}

// ---------------------------------------------------------------------------
// Kernel 1: fused char-CNN (conv as WMMA GEMM -> max over time -> tanh).
// One workgroup = 16 words. WMMA M dim = the 16 words, so max-over-time is an
// in-register elementwise max on identical C-fragment layouts.
// tanh is monotonic => max_p tanh(y_p+b) == tanh(max_p y_p + b); the hot loop
// is therefore only DS loads + WMMA + v_max_f32, with one tanh per feature.
// ---------------------------------------------------------------------------
template <int KSTEPS, int P>
__device__ __forceinline__ void conv_bank(
    const _Float16* __restrict__ wt, const float* __restrict__ bias,
    const _Float16* xs, int wg, int n0, int chbase, int lrow, int hi,
    float* __restrict__ feat32, _Float16* __restrict__ feat16) {
    const int Kpad = KSTEPS * 32;
    float bv = bias[n0 + lrow];
    // B fragments (32x16): lane col = lrow, K base = hi*16, 16 contiguous halves
    v16h bfrag[KSTEPS];
    const _Float16* bp = wt + (size_t)(n0 + lrow) * Kpad + hi * 16;
#pragma unroll
    for (int k = 0; k < KSTEPS; ++k) bfrag[k] = *(const v16h*)(bp + k * 32);

    float mx[8];
#pragma unroll
    for (int v = 0; v < 8; ++v) mx[v] = -3.0e38f;

    const _Float16* abase = xs + lrow * WSTRIDE + hi * 16;
    for (int p = 0; p < P; ++p) {
        v8f c = {};
        const _Float16* ar = abase + p * 50;  // window at position p: contiguous
#pragma unroll
        for (int k = 0; k < KSTEPS; ++k) {
            v16h a = load_a16(ar + k * 32);
            c = __builtin_amdgcn_wmma_f32_16x16x32_f16(
                false, a, false, bfrag[k], (short)0, c, false, false);
        }
#pragma unroll
        for (int v = 0; v < 8; ++v) mx[v] = fmaxf(mx[v], c[v]);
    }
    // C layout: row = v + hi*8, col = lrow
    int ch = chbase + n0 + lrow;
    size_t rowbase = (size_t)wg * 16 + hi * 8;
#pragma unroll
    for (int v = 0; v < 8; ++v) {
        size_t r = rowbase + v;
        float val = tanhf(mx[v] + bv);  // hoisted: bias + tanh once per feature
        feat32[r * OUT_DIM + ch] = val;
        feat16[r * OUT_DIM + ch] = (_Float16)val;
    }
}

__global__ void __launch_bounds__(256) charcnn_kernel(
    const float* __restrict__ x,  // (4096, 20*50) = (word, l*50+c)
    const _Float16* __restrict__ wt0, const _Float16* __restrict__ wt1,
    const _Float16* __restrict__ wt2, const float* __restrict__ b0,
    const float* __restrict__ b1, const float* __restrict__ b2,
    float* __restrict__ feat32, _Float16* __restrict__ feat16) {
    __shared__ _Float16 xs[16 * WSTRIDE];
    const int tid = threadIdx.x;
    const int wg = blockIdx.x;

    // stage 16 words (fp32 -> f16) + zero tail so padded-K reads are defined
    const float* xin = x + (size_t)wg * 16 * 1000;
    for (int idx = tid; idx < 16 * 1000; idx += 256) {
        int w = idx / 1000, j = idx - w * 1000;
        xs[w * WSTRIDE + j] = (_Float16)xin[idx];
    }
    for (int idx = tid; idx < 16 * (WSTRIDE - 1000); idx += 256) {
        int w = idx / (WSTRIDE - 1000), j = idx % (WSTRIDE - 1000);
        xs[w * WSTRIDE + 1000 + j] = (_Float16)0.0f;
    }
    __syncthreads();

    const int wave = tid >> 5, lane = tid & 31;
    const int lrow = lane & 15, hi = (lane >> 4) & 1;

    // 48 N-tiles total: bank0 [0,8), bank1 [8,16), bank2 [16,48)
    for (int t = wave; t < 48; t += 8) {
        if (t < 8)
            conv_bank<2, 20>(wt0, b0, xs, wg, t * 16, 0, lrow, hi, feat32, feat16);
        else if (t < 16)
            conv_bank<4, 19>(wt1, b1, xs, wg, (t - 8) * 16, 128, lrow, hi, feat32, feat16);
        else
            conv_bank<5, 18>(wt2, b2, xs, wg, (t - 16) * 16, 256, lrow, hi, feat32, feat16);
    }
}

// ---------------------------------------------------------------------------
// Kernel 2: highway layer. Block tile 64M x 64N, 8 waves as 4M x 2N; each wave
// does 2 N-tiles x 2 weight matrices (A fragment reused 4x). A staged in LDS.
// ---------------------------------------------------------------------------
__global__ void __launch_bounds__(256) highway_kernel(
    const _Float16* __restrict__ feat16, const float* __restrict__ feat32,
    const _Float16* __restrict__ whw, const _Float16* __restrict__ wtw,
    const float* __restrict__ whb, const float* __restrict__ wtb,
    float* __restrict__ out) {
    __shared__ _Float16 as[64 * 32];  // 64 rows x 32 K halves
    const int tid = threadIdx.x;
    const int wave = tid >> 5, lane = tid & 31;
    const int wr = wave >> 1, wc = wave & 1;
    const int lrow = lane & 15, hi = (lane >> 4) & 1;

    const size_t m0 = (size_t)blockIdx.x * 64;
    const int n0 = blockIdx.y * 64 + wc * 32;

    v8f ah0 = {}, ah1 = {}, at0 = {}, at1 = {};
    const _Float16* bh0 = whw + (size_t)(n0 + lrow) * OUT_DIM + hi * 16;
    const _Float16* bh1 = whw + (size_t)(n0 + 16 + lrow) * OUT_DIM + hi * 16;
    const _Float16* bt0 = wtw + (size_t)(n0 + lrow) * OUT_DIM + hi * 16;
    const _Float16* bt1 = wtw + (size_t)(n0 + 16 + lrow) * OUT_DIM + hi * 16;
    const _Float16* arow = as + (wr * 16 + lrow) * 32 + hi * 16;

    const int srow = tid >> 2, scg = tid & 3;  // staging: 4 x 16B per row
    for (int k0 = 0; k0 < OUT_DIM; k0 += 32) {
        __syncthreads();
        *(v8h*)(as + srow * 32 + scg * 8) =
            *(const v8h*)(feat16 + (m0 + srow) * OUT_DIM + k0 + scg * 8);
        __syncthreads();
        v16h a = load_a16(arow);
        ah0 = __builtin_amdgcn_wmma_f32_16x16x32_f16(false, a, false,
              *(const v16h*)(bh0 + k0), (short)0, ah0, false, false);
        ah1 = __builtin_amdgcn_wmma_f32_16x16x32_f16(false, a, false,
              *(const v16h*)(bh1 + k0), (short)0, ah1, false, false);
        at0 = __builtin_amdgcn_wmma_f32_16x16x32_f16(false, a, false,
              *(const v16h*)(bt0 + k0), (short)0, at0, false, false);
        at1 = __builtin_amdgcn_wmma_f32_16x16x32_f16(false, a, false,
              *(const v16h*)(bt1 + k0), (short)0, at1, false, false);
    }

    const int o0 = n0 + lrow, o1 = n0 + 16 + lrow;
    const float hb0 = whb[o0], hb1 = whb[o1];
    const float tb0 = wtb[o0], tb1 = wtb[o1];
    const size_t mbase = m0 + wr * 16 + hi * 8;
#pragma unroll
    for (int v = 0; v < 8; ++v) {
        size_t r = mbase + v;
        float f0 = feat32[r * OUT_DIM + o0];
        float f1 = feat32[r * OUT_DIM + o1];
        float h0 = fmaxf(ah0[v] + hb0, 0.0f);
        float h1 = fmaxf(ah1[v] + hb1, 0.0f);
        float t0 = 1.0f / (1.0f + expf(-(at0[v] + tb0)));
        float t1 = 1.0f / (1.0f + expf(-(at1[v] + tb1)));
        out[r * OUT_DIM + o0] = t0 * h0 + (1.0f - t0) * f0;
        out[r * OUT_DIM + o1] = t1 * h1 + (1.0f - t1) * f1;
    }
}

// ---------------------------------------------------------------------------
// Launch. Workspace layout (bytes, all 256B-aligned; ~20.5 MB total):
//   wt0 f16 @0        (16384)    wt1 f16 @16384    (32768)
//   wt2 f16 @49152    (163840)   whw f16 @212992   (1179648)
//   wtw f16 @1392640  (1179648)  feat32  @2572288  (12582912)
//   feat16  @15155200 (6291456)
// ---------------------------------------------------------------------------
extern "C" void kernel_launch(void* const* d_in, const int* in_sizes, int n_in,
                              void* d_out, int out_size, void* d_ws, size_t ws_size,
                              hipStream_t stream) {
    const float* x   = (const float*)d_in[0];
    const float* cw0 = (const float*)d_in[1];
    const float* cb0 = (const float*)d_in[2];
    const float* cw1 = (const float*)d_in[3];
    const float* cb1 = (const float*)d_in[4];
    const float* cw2 = (const float*)d_in[5];
    const float* cb2 = (const float*)d_in[6];
    const float* whw = (const float*)d_in[7];
    const float* whb = (const float*)d_in[8];
    const float* wtw = (const float*)d_in[9];
    const float* wtb = (const float*)d_in[10];
    float* out = (float*)d_out;

    char* ws = (char*)d_ws;
    _Float16* wt0    = (_Float16*)(ws + 0);
    _Float16* wt1    = (_Float16*)(ws + 16384);
    _Float16* wt2    = (_Float16*)(ws + 49152);
    _Float16* whw16  = (_Float16*)(ws + 212992);
    _Float16* wtw16  = (_Float16*)(ws + 1392640);
    float*    feat32 = (float*)   (ws + 2572288);
    _Float16* feat16 = (_Float16*)(ws + 15155200);

    prep_kernel<<<512, 256, 0, stream>>>(cw0, cw1, cw2, whw, wtw,
                                         wt0, wt1, wt2, whw16, wtw16);
    charcnn_kernel<<<NWORDS / 16, 256, 0, stream>>>(x, wt0, wt1, wt2,
                                                    cb0, cb1, cb2, feat32, feat16);
    highway_kernel<<<dim3(NWORDS / 64, OUT_DIM / 64), 256, 0, stream>>>(
        feat16, feat32, whw16, wtw16, whb, wtb, out);
}
